// CoreKT_21131239096613
// MI455X (gfx1250) — matrix operand order repaired and checked by reading, not compiled, not run on hardware
//
#include <hip/hip_runtime.h>
#include <math.h>

#define BB   8
#define SS   1024
#define EE   256
#define HH   8
#define DKK  32
#define DFFN 1024
#define MTOT (BB * SS)          // 8192 rows
#define NE   (BB * SS * EE)     // 2097152 elements per [B,S,E] tensor

typedef __attribute__((ext_vector_type(16))) __bf16 v16bf;
typedef __attribute__((ext_vector_type(8)))  __bf16 v8bf;
typedef __attribute__((ext_vector_type(8)))  float  v8f;

// ---------------------------------------------------------------- helpers ---

__device__ __forceinline__ __bf16 f2bf(float f) {
    unsigned u = __builtin_bit_cast(unsigned, f);
    unsigned r = (u + 0x7FFFu + ((u >> 16) & 1u)) >> 16;   // RNE
    unsigned short h = (unsigned short)r;
    return __builtin_bit_cast(__bf16, h);
}

// pack 16 fp32 (four float4, in fragment order) into a v16bf
__device__ __forceinline__ v16bf pack16(const float4& a0, const float4& a1,
                                        const float4& a2, const float4& a3) {
    v16bf v;
    v[0]  = f2bf(a0.x); v[1]  = f2bf(a0.y); v[2]  = f2bf(a0.z); v[3]  = f2bf(a0.w);
    v[4]  = f2bf(a1.x); v[5]  = f2bf(a1.y); v[6]  = f2bf(a1.z); v[7]  = f2bf(a1.w);
    v[8]  = f2bf(a2.x); v[9]  = f2bf(a2.y); v[10] = f2bf(a2.z); v[11] = f2bf(a2.w);
    v[12] = f2bf(a3.x); v[13] = f2bf(a3.y); v[14] = f2bf(a3.z); v[15] = f2bf(a3.w);
    return v;
}

__device__ __forceinline__ float wave_sum(float x) {
    #pragma unroll
    for (int off = 16; off > 0; off >>= 1) x += __shfl_xor(x, off, 32);
    return x;
}
__device__ __forceinline__ float wave_max(float x) {
    #pragma unroll
    for (int off = 16; off > 0; off >>= 1) x = fmaxf(x, __shfl_xor(x, off, 32));
    return x;
}

// ------------------------------------------------------------------- GEMM ---
// Y[M,N] = act(X[M,K] @ W[K,N] + bias[N]).
// 128 threads = 4 waves; block tile 64(M) x 32(N); each wave: 16x32 (2 WMMAs).
// A staged via LDS (bf16, b128 reads); B read directly from global (L2-resident
// weights, 64B-coalesced per h); next A k-slice prefetched.
__global__ __launch_bounds__(128) void gemm_bias(
    const float* __restrict__ X, const float* __restrict__ W,
    const float* __restrict__ bias, float* __restrict__ Y,
    int M, int K, int N, int relu)
{
    __shared__ __bf16 As[64][40];                   // 80B row pitch (16B multiple)
    const int t    = threadIdx.x;
    const int lane = t & 31, wv = t >> 5;
    const int m0   = blockIdx.y * 64, n0 = blockIdx.x * 32;
    const int r    = lane & 15, kh = lane >> 4;

    v8f acc[2] = {{}, {}};
    for (int k0 = 0; k0 < K; k0 += 32) {
        #pragma unroll
        for (int i = 0; i < 4; ++i) {               // stage 64x32 A -> bf16 LDS
            int f4 = t + i * 128;
            int row = f4 >> 3, c4 = (f4 & 7) * 4;
            const float* src = X + (size_t)(m0 + row) * K + k0 + c4;
            float4 av = *(const float4*)src;
            As[row][c4 + 0] = f2bf(av.x); As[row][c4 + 1] = f2bf(av.y);
            As[row][c4 + 2] = f2bf(av.z); As[row][c4 + 3] = f2bf(av.w);
            if (k0 + 32 < K) __builtin_prefetch(src + 32, 0, 3);
        }
        __syncthreads();
        // A fragment: two contiguous 8x bf16 runs -> 2x ds_load_b128
        const __bf16* arow = &As[wv * 16 + r][0];
        v8bf lo = *(const v8bf*)(arow + 16 * kh);
        v8bf hi = *(const v8bf*)(arow + 16 * kh + 8);
        v16bf a = __builtin_shufflevector(lo, hi, 0, 1, 2, 3, 4, 5, 6, 7,
                                          8, 9, 10, 11, 12, 13, 14, 15);
        #pragma unroll
        for (int nh = 0; nh < 2; ++nh) {
            int n = n0 + nh * 16 + r;
            v16bf b;
            #pragma unroll
            for (int h = 0; h < 16; ++h)            // B 32x16 layout: K = h+16*kh
                b[h] = f2bf(W[(size_t)(k0 + h + 16 * kh) * N + n]);
            acc[nh] = __builtin_amdgcn_wmma_f32_16x16x32_bf16(
                false, a, false, b, (short)0, acc[nh], false, false);
        }
        __syncthreads();
    }
    #pragma unroll
    for (int nh = 0; nh < 2; ++nh) {
        int n = n0 + nh * 16 + r;
        float bb = bias[n];
        #pragma unroll
        for (int i = 0; i < 8; ++i) {
            int m = m0 + wv * 16 + i + 8 * kh;      // C/D layout
            float v = acc[nh][i] + bb;
            if (relu) v = fmaxf(v, 0.f);
            Y[(size_t)m * N + n] = v;
        }
    }
}

// -------------------------------------------------------------- attention ---
// One workgroup per (qtile=16 rows, head, batch). 64KB dynamic LDS scores.
__global__ __launch_bounds__(128) void attn_kernel(
    const float* __restrict__ pKQ, const float* __restrict__ pV,
    const float* __restrict__ gammas, float* __restrict__ ctx, int mask_k)
{
    extern __shared__ float sc[];                   // [16][SS]
    const int t = threadIdx.x, lane = t & 31, wv = t >> 5;
    const int r = lane & 15, kh = lane >> 4;
    const int q0 = blockIdx.x * 16, head = blockIdx.y, b = blockIdx.z;
    const int col0 = head * DKK;
    const size_t baseRow = (size_t)b * SS;
    const float scale = 0.17677669529663687f;       // 1/sqrt(32)

    // ---- phase 1: scores = q @ k^T * scale  (K-dim = DK = 32 -> 1 WMMA/tile)
    const float* qrow = pKQ + (baseRow + q0 + r) * EE + col0;
    v16bf aq = pack16(*(const float4*)(qrow + 16 * kh),
                      *(const float4*)(qrow + 16 * kh + 4),
                      *(const float4*)(qrow + 16 * kh + 8),
                      *(const float4*)(qrow + 16 * kh + 12));
    for (int tile = wv; tile < SS / 16; tile += 4) {
        int key0 = tile * 16;
        const float* krow = pKQ + (baseRow + key0 + r) * EE + col0 + 16 * kh;
        v16bf bk = pack16(*(const float4*)(krow + 0),  *(const float4*)(krow + 4),
                          *(const float4*)(krow + 8),  *(const float4*)(krow + 12));
        v8f d = {};
        d = __builtin_amdgcn_wmma_f32_16x16x32_bf16(false, aq, false, bk,
                                                    (short)0, d, false, false);
        #pragma unroll
        for (int i = 0; i < 8; ++i)
            sc[(i + 8 * kh) * SS + key0 + r] = d[i] * scale;
    }
    __syncthreads();

    // ---- phase 2: per-row masked softmax / cumsum decay / second softmax
    float g  = gammas[head];
    float sp = (g > 0.f) ? (g + log1pf(__expf(-g))) : log1pf(__expf(g));
    float gamma = -sp;                              // -softplus(gamma_h)

    for (int rr = wv; rr < 16; rr += 4) {
        float* row = sc + rr * SS;
        int i = q0 + rr;
        int limit = i + mask_k - 1;                 // valid: j <= limit
        if (limit < 0) {                            // fully masked row -> zero_pad
            for (int c = 0; c < 32; ++c) row[c * 32 + lane] = 0.f;
            continue;
        }
        float rv[32], cum[32];
        float mx = -3.0e38f;
        #pragma unroll
        for (int c = 0; c < 32; ++c) {
            int j = c * 32 + lane;
            rv[c] = row[j];
            if (j <= limit) mx = fmaxf(mx, rv[c]);
        }
        mx = wave_max(mx);
        float se = 0.f;
        #pragma unroll
        for (int c = 0; c < 32; ++c) {
            int j = c * 32 + lane;
            se += (j <= limit) ? __expf(rv[c] - mx) : 0.f;
        }
        se = wave_sum(se);
        float inv = 1.f / se;
        float carry = 0.f;                          // inclusive cumsum of s1
        #pragma unroll
        for (int c = 0; c < 32; ++c) {
            int j = c * 32 + lane;
            float xs = (j <= limit) ? __expf(rv[c] - mx) * inv : 0.f;
            for (int off = 1; off < 32; off <<= 1) {
                float tv = __shfl_up(xs, off, 32);
                if (lane >= off) xs += tv;
            }
            cum[c] = carry + xs;
            carry += __shfl(xs, 31, 32);
        }
        float disttot = carry;
        float mx2 = -3.0e38f;
        #pragma unroll
        for (int c = 0; c < 32; ++c) {
            int j = c * 32 + lane;
            float pos  = fabsf((float)(i - j));
            float dist = sqrtf(fmaxf((disttot - cum[c]) * pos, 0.f));
            float eff  = fminf(fmaxf(__expf(dist * gamma), 1e-5f), 1e5f);
            float s2   = (j <= limit) ? rv[c] * eff : -3.0e38f;
            rv[c] = s2;
            if (j <= limit) mx2 = fmaxf(mx2, s2);
        }
        mx2 = wave_max(mx2);
        float s2sum = 0.f;
        #pragma unroll
        for (int c = 0; c < 32; ++c) {
            int j = c * 32 + lane;
            float e = (j <= limit) ? __expf(rv[c] - mx2) : 0.f;
            rv[c] = e;
            s2sum += e;
        }
        s2sum = wave_sum(s2sum);
        float inv2 = 1.f / s2sum;
        #pragma unroll
        for (int c = 0; c < 32; ++c) row[c * 32 + lane] = rv[c] * inv2;
    }
    __syncthreads();

    // ---- phase 3: ctx = P @ V   (waves 0/1 handle the two 16-wide DK halves)
    if (wv < 2) {
        int nh = wv;
        v8f acc = {};
        for (int tile = 0; tile < 32; ++tile) {
            int key0 = tile * 32;
            const float* prow = sc + r * SS + key0; // LDS, b128 reads
            v16bf ap = pack16(*(const float4*)(prow + 16 * kh),
                              *(const float4*)(prow + 16 * kh + 4),
                              *(const float4*)(prow + 16 * kh + 8),
                              *(const float4*)(prow + 16 * kh + 12));
            v16bf bv;
            #pragma unroll
            for (int h = 0; h < 16; ++h) {
                int key = key0 + h + 16 * kh;
                bv[h] = f2bf(pV[(baseRow + key) * EE + col0 + nh * 16 + r]);
            }
            acc = __builtin_amdgcn_wmma_f32_16x16x32_bf16(false, ap, false, bv,
                                                          (short)0, acc, false, false);
        }
        int f = col0 + nh * 16 + r;                 // concat-head layout
        #pragma unroll
        for (int i2 = 0; i2 < 8; ++i2)
            ctx[(baseRow + q0 + i2 + 8 * kh) * EE + f] = acc[i2];
    }
}

// ------------------------------------------------------- residual + LN ------
__global__ __launch_bounds__(128) void add_ln(
    float* __restrict__ x, const float* __restrict__ q2,
    const float* __restrict__ g, const float* __restrict__ bta)
{
    int lane = threadIdx.x & 31, wv = threadIdx.x >> 5;
    size_t row = (size_t)blockIdx.x * 4 + wv;
    float* xr = x + row * EE + lane * 8;            // contiguous 8 per lane
    const float* qr = q2 + row * EE + lane * 8;
    float4 a0 = *(const float4*)xr,       a1 = *(const float4*)(xr + 4);
    float4 b0 = *(const float4*)qr,       b1 = *(const float4*)(qr + 4);
    float v[8] = {a0.x + b0.x, a0.y + b0.y, a0.z + b0.z, a0.w + b0.w,
                  a1.x + b1.x, a1.y + b1.y, a1.z + b1.z, a1.w + b1.w};
    float s = 0.f;
    #pragma unroll
    for (int e = 0; e < 8; ++e) s += v[e];
    s = wave_sum(s);
    float mu = s * (1.f / EE);
    float s2 = 0.f;
    #pragma unroll
    for (int e = 0; e < 8; ++e) { float d = v[e] - mu; s2 += d * d; }
    s2 = wave_sum(s2);
    float rs = rsqrtf(s2 * (1.f / EE) + 1e-5f);
    float4 g0 = *(const float4*)(g + lane * 8),   g1 = *(const float4*)(g + lane * 8 + 4);
    float4 t0 = *(const float4*)(bta + lane * 8), t1 = *(const float4*)(bta + lane * 8 + 4);
    float4 o0, o1;
    o0.x = (v[0] - mu) * rs * g0.x + t0.x;  o0.y = (v[1] - mu) * rs * g0.y + t0.y;
    o0.z = (v[2] - mu) * rs * g0.z + t0.z;  o0.w = (v[3] - mu) * rs * g0.w + t0.w;
    o1.x = (v[4] - mu) * rs * g1.x + t1.x;  o1.y = (v[5] - mu) * rs * g1.y + t1.y;
    o1.z = (v[6] - mu) * rs * g1.z + t1.z;  o1.w = (v[7] - mu) * rs * g1.w + t1.w;
    *(float4*)xr = o0;
    *(float4*)(xr + 4) = o1;
}

// ------------------------------------------------------------------- host ---

struct LayerP {
    const float *kw, *kb, *vw, *vb, *ow, *ob, *gam;
    const float *l1w, *l1b, *l2w, *l2b, *ln1g, *ln1b, *ln2g, *ln2b;
};

static LayerP get_layer(void* const* d_in, int p0, int li, bool alpha) {
    const int base = p0 + li * 15;
    auto P = [&](int o) { return (const float*)d_in[base + o]; };
    LayerP L;
    if (alpha) {   // jax tree_leaves order (dict keys sorted)
        L.gam = P(0);  L.kb = P(1);   L.kw = P(2);   L.l1b = P(3);  L.l1w = P(4);
        L.l2b = P(5);  L.l2w = P(6);  L.ln1b = P(7); L.ln1g = P(8); L.ln2b = P(9);
        L.ln2g = P(10); L.ob = P(11); L.ow = P(12);  L.vb = P(13);  L.vw = P(14);
    } else {       // python dict insertion order
        L.kw = P(0);   L.kb = P(1);   L.vw = P(2);   L.vb = P(3);   L.ow = P(4);
        L.ob = P(5);   L.gam = P(6);  L.ln1g = P(7); L.ln1b = P(8); L.l1w = P(9);
        L.l1b = P(10); L.l2w = P(11); L.l2b = P(12); L.ln2g = P(13); L.ln2b = P(14);
    }
    return L;
}

static void run_layer(float* cur, const float* vsrc, const LayerP& L, int mask_k,
                      bool apply_pos, float* projKQ, float* projV, float* ctx,
                      float* q2, float* hid, hipStream_t stream)
{
    dim3 blk(128);
    gemm_bias<<<dim3(EE / 32, MTOT / 64), blk, 0, stream>>>(cur,  L.kw, L.kb, projKQ, MTOT, EE, EE, 0);
    gemm_bias<<<dim3(EE / 32, MTOT / 64), blk, 0, stream>>>(vsrc, L.vw, L.vb, projV,  MTOT, EE, EE, 0);
    attn_kernel<<<dim3(SS / 16, HH, BB), blk, 16 * SS * sizeof(float), stream>>>(
        projKQ, projV, L.gam, ctx, mask_k);
    gemm_bias<<<dim3(EE / 32, MTOT / 64), blk, 0, stream>>>(ctx, L.ow, L.ob, q2, MTOT, EE, EE, 0);
    add_ln<<<dim3(MTOT / 4), blk, 0, stream>>>(cur, q2, L.ln1g, L.ln1b);
    if (apply_pos) {
        gemm_bias<<<dim3(DFFN / 32, MTOT / 64), blk, 0, stream>>>(cur, L.l1w, L.l1b, hid, MTOT, EE, DFFN, 1);
        gemm_bias<<<dim3(EE / 32, MTOT / 64), blk, 0, stream>>>(hid, L.l2w, L.l2b, q2, MTOT, DFFN, EE, 0);
        add_ln<<<dim3(MTOT / 4), blk, 0, stream>>>(cur, q2, L.ln2g, L.ln2b);
    }
}

extern "C" void kernel_launch(void* const* d_in, const int* in_sizes, int n_in,
                              void* d_out, int out_size, void* d_ws, size_t ws_size,
                              hipStream_t stream)
{
    (void)out_size; (void)ws_size;
    // Locate embeddings vs. flattened params (handles both flattening styles).
    int qIdx = 0, qaIdx = 1, p0 = 2;
    if (n_in >= 92 && in_sizes[0] != NE) { p0 = 0; qIdx = n_in - 2; qaIdx = n_in - 1; }
    const bool alpha = (in_sizes[p0] == 8);         // first leaf == gammas(8) -> sorted order

    const float* q_emb  = (const float*)d_in[qIdx];
    const float* qa_emb = (const float*)d_in[qaIdx];

    float* xbuf = (float*)d_out;                    // output x [B,S,E]
    float* ybuf = (float*)d_out + NE;               // output y [B,S,E]

    float* ws      = (float*)d_ws;
    float* projKQ  = ws;                            // NE
    float* projV   = ws + (size_t)NE;               // NE
    float* ctx     = ws + (size_t)2 * NE;           // NE
    float* q2      = ws + (size_t)3 * NE;           // NE
    float* hid     = ws + (size_t)4 * NE;           // 4*NE  ([B,S,DFF])

    hipMemcpyAsync(ybuf, qa_emb, (size_t)NE * sizeof(float), hipMemcpyDeviceToDevice, stream);
    hipMemcpyAsync(xbuf, q_emb,  (size_t)NE * sizeof(float), hipMemcpyDeviceToDevice, stream);

    // y-chain: blocks_1 (mask_k=1, FFN)
    for (int i = 0; i < 2; ++i) {
        LayerP L = get_layer(d_in, p0, i, alpha);
        run_layer(ybuf, ybuf, L, 1, true, projKQ, projV, ctx, q2, hid, stream);
    }
    // x-chain: blocks_2 alternating (mask 1 / no FFN) and (mask 0 / values=y / FFN)
    {
        LayerP L = get_layer(d_in, p0, 2, alpha);
        run_layer(xbuf, xbuf, L, 1, false, projKQ, projV, ctx, q2, hid, stream);
        L = get_layer(d_in, p0, 3, alpha);
        run_layer(xbuf, ybuf, L, 0, true,  projKQ, projV, ctx, q2, hid, stream);
        L = get_layer(d_in, p0, 4, alpha);
        run_layer(xbuf, xbuf, L, 1, false, projKQ, projV, ctx, q2, hid, stream);
        L = get_layer(d_in, p0, 5, alpha);
        run_layer(xbuf, ybuf, L, 0, true,  projKQ, projV, ctx, q2, hid, stream);
    }
}
